// RNN_21406117003943
// MI455X (gfx1250) — matrix operand order repaired and checked by reading, not compiled, not run on hardware
//
#include <hip/hip_runtime.h>
#include <cstddef>

// ---------------------------------------------------------------------------
// RNN (B=256, T=1024, EMBED=256, HIDDEN=512, OUT=256), fp32 reference.
// Strategy: batch-parallel persistent workgroups (16 WGs x 16 rows), bf16
// WMMA (v_wmma_f32_16x16x32_bf16) with f32 accumulators, Wd pinned in LDS,
// W streamed from L2 in pre-swizzled WMMA B-tile layout, projection fused
// per timestep (hs never materialized).
//
// 512 threads = 16 wave32 -> 4 waves/SIMD on the single WGP-resident WG:
// same WMMA count per SIMD per step, 2x the independent load streams to
// hide L2 latency on the streamed recurrence weights.
// ---------------------------------------------------------------------------

typedef __attribute__((ext_vector_type(16))) __bf16 v16bf;
typedef __attribute__((ext_vector_type(8)))  __bf16 v8bf;
typedef __attribute__((ext_vector_type(8)))  float  v8f;

constexpr int B_      = 256;
constexpr int T_      = 1024;
constexpr int EMBED_  = 256;
constexpr int HIDDEN_ = 512;
constexpr int OUT_    = 256;
constexpr int KTOT_   = EMBED_ + HIDDEN_;       // 768
constexpr int WG_ROWS = 16;                     // batch rows per workgroup
constexpr int NBLOCKS = B_ / WG_ROWS;           // 16 workgroups
constexpr int NTHREADS = 512;                   // 16 waves

// LDS layout: Wd packed tiles (256 KB) + A buffer [e_t | h] bf16 (padded)
constexpr int A_STRIDE    = KTOT_ + 8;          // 776 halves -> 4-bank row shift, conflict-free b128
constexpr int SWD_HALVES  = (HIDDEN_ / 32) * (OUT_ / 16) * 512;  // 131072
constexpr int SMEM_BYTES  = SWD_HALVES * 2 + WG_ROWS * A_STRIDE * 2;  // 286976 < 320KB

__device__ __forceinline__ v16bf cat16(v8bf lo, v8bf hi) {
  return __builtin_shufflevector(lo, hi, 0,1,2,3,4,5,6,7,8,9,10,11,12,13,14,15);
}
__device__ __forceinline__ v8f wmma_bf16(v16bf a, v16bf bm, v8f c) {
  // (neg_a, A, neg_b, B, c_mod, C, reuse_a, reuse_b)
  return __builtin_amdgcn_wmma_f32_16x16x32_bf16(false, a, false, bm, (short)0, c, false, false);
}

// ---------------------------------------------------------------------------
// Prep: f32 -> bf16, and swizzle W / Wd into WMMA B-tile layout:
//   packed[(kt*NT + nt)*512 + lane*16 + j] = W[n = nt*16 + (lane&15)]
//                                              [k = kt*32 + 16*(lane>>4) + j]
// so each lane's 16 K-halves are 32 contiguous bytes (2x b128, fully coalesced).
// ---------------------------------------------------------------------------
__global__ void prep_kernel(const float* __restrict__ emb, const float* __restrict__ W,
                            const float* __restrict__ Wd, __bf16* __restrict__ embBF,
                            __bf16* __restrict__ Wp, __bf16* __restrict__ Wdp)
{
  int i = blockIdx.x * blockDim.x + threadIdx.x;
  if (i < EMBED_ * 256) embBF[i] = (__bf16)emb[i];               // VOCAB x EMBED
  if (i < KTOT_ * HIDDEN_) {                                     // W: (512 x 768), 24x32 tiles
    int tile = i >> 9, w = i & 511, lane = w >> 4, j = w & 15;
    int kt = tile >> 5, nt = tile & 31;
    int n = nt * 16 + (lane & 15);
    int k = kt * 32 + ((lane >> 4) << 4) + j;
    Wp[i] = (__bf16)W[n * KTOT_ + k];
  }
  if (i < HIDDEN_ * OUT_) {                                      // Wd: (256 x 512), 16x16 tiles
    int tile = i >> 9, w = i & 511, lane = w >> 4, j = w & 15;
    int kt = tile >> 4, nt = tile & 15;
    int n = nt * 16 + (lane & 15);
    int k = kt * 32 + ((lane >> 4) << 4) + j;
    Wdp[i] = (__bf16)Wd[n * HIDDEN_ + k];
  }
}

// ---------------------------------------------------------------------------
// Persistent recurrent kernel: one WG = 16 batch rows, 16 waves (wave32).
// Recurrence: wave w owns hidden N-tiles {2w, 2w+1}  (columns [w*32, w*32+32)).
// Projection: wave w owns out   N-tile   w          (columns [w*16, w*16+16)).
// ---------------------------------------------------------------------------
__global__ void __launch_bounds__(NTHREADS, 1)
rnn_fused(const int* __restrict__ x, const float* __restrict__ h0,
          const float* __restrict__ b, const float* __restrict__ bd,
          const __bf16* __restrict__ embBF, const __bf16* __restrict__ Wp,
          const __bf16* __restrict__ Wdp, float* __restrict__ out,
          float* __restrict__ hfin)
{
  extern __shared__ char smem[];
  __bf16* sWd = (__bf16*)smem;                         // packed Wd tiles, 256 KB
  __bf16* sA  = (__bf16*)(smem + SWD_HALVES * 2);      // [16][A_STRIDE] bf16: cols 0..255 = e_t, 256..767 = h

  const int tid    = (int)threadIdx.x;
  const int wave   = tid >> 5;
  const int lane   = tid & 31;
  const int lrow   = lane & 15;     // A-row / B-column / C-column within a tile
  const int lhi    = lane >> 4;     // K half-group selector
  const int wgbase = (int)blockIdx.x * WG_ROWS;

  // Stage packed Wd into LDS (b128 copies).
  {
    const v8bf* g = (const v8bf*)Wdp;
    v8bf*       s = (v8bf*)sWd;
    for (int i = tid; i < SWD_HALVES / 8; i += NTHREADS) s[i] = g[i];
  }
  // Stage initial h (f32 -> bf16) into the A buffer's h region.
  for (int i = tid; i < WG_ROWS * HIDDEN_; i += NTHREADS) {
    int m = i >> 9, n = i & 511;
    sA[m * A_STRIDE + EMBED_ + n] = (__bf16)h0[(wgbase + m) * HIDDEN_ + n];
  }
  __syncthreads();

  const int ntb = wave * 2;         // recurrence N-tile base (hidden, 2 tiles/wave)
  const int gr  = tid >> 5;         // gather: row 0..15
  const int gc  = tid & 31;         // gather: 8-half chunk 0..31

  for (int t = 0; t < T_; ++t) {
    // --- embedding gather into A[:, 0:256]: one b128 per thread (writes only
    //     the e-region: no hazard vs previous step's projection, which reads
    //     the h-region and sWd only)
    {
      int token = x[(wgbase + gr) * T_ + t];
      *(v8bf*)(sA + gr * A_STRIDE + gc * 8) =
          *(const v8bf*)(embBF + token * EMBED_ + gc * 8);
    }
    __syncthreads();

    // --- recurrence: C(16x512) += A(16x768) x B(768x512), K in 24 tiles of 32
    v8f acc0 = {}, acc1 = {};
    for (int kt = 0; kt < KTOT_ / 32; ++kt) {
      const __bf16* ap = sA + lrow * A_STRIDE + kt * 32 + lhi * 16;  // A layout: K = e + 16*lhi
      v16bf a = cat16(*(const v8bf*)ap, *(const v8bf*)(ap + 8));
      const v8bf* bp = (const v8bf*)(Wp + (size_t)(kt * 32 + ntb) * 512) + lane * 2;
      acc0 = wmma_bf16(a, cat16(bp[0],  bp[1]),  acc0);
      acc1 = wmma_bf16(a, cat16(bp[64], bp[65]), acc1);
    }
    __syncthreads();   // every wave finished reading A before h is overwritten

    // --- bias + tanh, write new h (bf16) back to A[:, 256:768]
    const bool last = (t == T_ - 1);
#pragma unroll
    for (int i = 0; i < 2; ++i) {
      v8f acc = (i == 0) ? acc0 : acc1;
      int n = (ntb + i) * 16 + lrow;
      float bias = b[n];
#pragma unroll
      for (int v = 0; v < 8; ++v) {
        float hval = tanhf(acc[v] + bias);       // C layout: M = v + 8*lhi, N = lrow
        int m = v + lhi * 8;
        sA[m * A_STRIDE + EMBED_ + n] = (__bf16)hval;
        if (last) hfin[(size_t)(wgbase + m) * HIDDEN_ + n] = hval;
      }
    }
    __syncthreads();   // new h visible to all waves

    // --- fused projection: out_t(16x256) = h(16x512) x Wd^T, Wd tiles from LDS
    v8f p0 = {};
    for (int kt = 0; kt < HIDDEN_ / 32; ++kt) {
      const __bf16* ap = sA + lrow * A_STRIDE + EMBED_ + kt * 32 + lhi * 16;
      v16bf a = cat16(*(const v8bf*)ap, *(const v8bf*)(ap + 8));
      const v8bf* bp = (const v8bf*)(sWd + (size_t)(kt * 16 + wave) * 512) + lane * 2;
      p0 = wmma_bf16(a, cat16(bp[0], bp[1]), p0);
    }
    {
      int o = wave * 16 + lrow;
      float bo = bd[o];
#pragma unroll
      for (int v = 0; v < 8; ++v) {
        int m = v + lhi * 8;
        out[((size_t)(wgbase + m) * T_ + t) * OUT_ + o] = p0[v] + bo;
      }
    }
  }
}

// ---------------------------------------------------------------------------
// Host launcher. Inputs (setup_inputs order): x, h, emb, W, b, Wd, bd.
// d_out = outputs (B,T,OUT) f32 ++ h_final (B,HIDDEN) f32.
// Workspace: embBF (128KB) | Wp packed (768KB) | Wdp packed (256KB) ~ 1.13 MB.
// ---------------------------------------------------------------------------
extern "C" void kernel_launch(void* const* d_in, const int* in_sizes, int n_in,
                              void* d_out, int out_size, void* d_ws, size_t ws_size,
                              hipStream_t stream) {
  (void)in_sizes; (void)n_in; (void)out_size; (void)ws_size;

  const int*   x   = (const int*)  d_in[0];
  const float* h0  = (const float*)d_in[1];
  const float* emb = (const float*)d_in[2];
  const float* W   = (const float*)d_in[3];
  const float* b   = (const float*)d_in[4];
  const float* Wd  = (const float*)d_in[5];
  const float* bd  = (const float*)d_in[6];
  float* out = (float*)d_out;

  __bf16* embBF = (__bf16*)d_ws;
  __bf16* Wp    = embBF + 256 * EMBED_;
  __bf16* Wdp   = Wp + KTOT_ * HIDDEN_;

  // Deterministic convert + swizzle (rerun every call).
  prep_kernel<<<(KTOT_ * HIDDEN_) / 256, 256, 0, stream>>>(emb, W, Wd, embBF, Wp, Wdp);

  (void)hipFuncSetAttribute(reinterpret_cast<const void*>(rnn_fused),
                            hipFuncAttributeMaxDynamicSharedMemorySize, SMEM_BYTES);

  float* hfin = out + (size_t)B_ * T_ * OUT_;
  rnn_fused<<<NBLOCKS, NTHREADS, SMEM_BYTES, stream>>>(x, h0, b, bd, embBF, Wp, Wdp, out, hfin);
}
